// Transformer_83777632076044
// MI455X (gfx1250) — matrix operand order repaired and checked by reading, not compiled, not run on hardware
//
#include <hip/hip_runtime.h>
#include <hip/hip_bf16.h>

// ---------------------------------------------------------------------------
// Model constants (match reference)
// ---------------------------------------------------------------------------
static constexpr int N0V   = 100000;          // N_RAW
static constexpr int NPAD  = 100096;          // padded to BLOCK=128
static constexpr int NS    = 131072;          // pow2 pad for bitonic sort
static constexpr int NBLK  = NPAD / 128;      // 782 attention blocks
static constexpr int HEADS = 4;
static constexpr int HD    = 64;              // dim per head
static constexpr int HDIM  = 256;             // HEADS*HD
static constexpr int LAYERS= 2;
static constexpr int NHASH = 2;
static constexpr int INDIM = 32;
static constexpr int ENCD  = 192;             // (LAYERS+1)*HD
static constexpr int D2    = 32;
static constexpr int MLPH  = 256;

typedef __attribute__((ext_vector_type(16))) _Float16 v16h;
typedef __attribute__((ext_vector_type(8)))  float    v8f;

// ---------------------------------------------------------------------------
// Sorting: order-preserving keys + bitonic network (stable via idx-in-low-bits)
// ---------------------------------------------------------------------------
__device__ __forceinline__ unsigned f2ord(float f) {
  unsigned u = __float_as_uint(f);
  return (u & 0x80000000u) ? ~u : (u | 0x80000000u);
}

__global__ void k_keys_coords(const float* __restrict__ coords,
                              unsigned long long* __restrict__ keys) {
  int i = blockIdx.x * blockDim.x + threadIdx.x;   // 0 .. 2*NS-1
  if (i >= 2 * NS) return;
  int c = i / NS, n = i - c * NS;
  float v = (n < N0V) ? coords[(size_t)n * 2 + c] : __builtin_inff(); // pad rows -> +inf
  keys[(size_t)c * NS + n] = ((unsigned long long)f2ord(v) << 32) | (unsigned)n;
}

__global__ void k_bitonic64(unsigned long long* __restrict__ keys, int kk, int jj) {
  unsigned long long* a = keys + (size_t)blockIdx.y * NS;
  int i = blockIdx.x * blockDim.x + threadIdx.x;
  if (i >= NS) return;
  int p = i ^ jj;
  if (p > i) {
    bool up = ((i & kk) == 0);
    unsigned long long x = a[i], y = a[p];
    if (up ? (x > y) : (x < y)) { a[i] = y; a[p] = x; }
  }
}

__global__ void k_bitonic32(unsigned* __restrict__ keys, int kk, int jj) {
  unsigned* a = keys + (size_t)blockIdx.y * NS;
  int i = blockIdx.x * blockDim.x + threadIdx.x;
  if (i >= NS) return;
  int p = i ^ jj;
  if (p > i) {
    bool up = ((i & kk) == 0);
    unsigned x = a[i], y = a[p];
    if (up ? (x > y) : (x < y)) { a[i] = y; a[p] = x; }
  }
}

__global__ void k_ranks(const unsigned long long* __restrict__ keys,
                        float* __restrict__ ranks) {
  int i = blockIdx.x * blockDim.x + threadIdx.x;   // 0 .. 2*NS-1
  if (i >= 2 * NS) return;
  int c = i / NS, pos = i - c * NS;
  unsigned idx = (unsigned)(keys[(size_t)c * NS + pos] & 0xFFFFFFFFull);
  if (idx < (unsigned)NPAD) ranks[(size_t)c * NPAD + idx] = (float)(pos + 1);
}

// comb key = region_id << 17 | idx   (comb <= 336, idx < 2^17)
__global__ void k_combkeys(const float* __restrict__ ranks,
                           const float* __restrict__ regions,  // [NHASH][2][HEADS]
                           unsigned* __restrict__ keys) {
  int i = blockIdx.x * blockDim.x + threadIdx.x;   // 0 .. 8*NS-1
  if (i >= NHASH * HEADS * NS) return;
  int a = i / NS, n = i - a * NS;
  if (n >= NPAD) { keys[(size_t)a * NS + n] = 0xFFFFFFFFu; return; }
  int hsh = a / HEADS, hd = a % HEADS;
  float re = regions[(size_t)hsh * 2 * HEADS + 0 * HEADS + hd];
  float rp = regions[(size_t)hsh * 2 * HEADS + 1 * HEADS + hd];
  float mx = 0.f;
  for (int t = 0; t < NHASH * HEADS; t++)
    mx = fmaxf(mx, regions[(size_t)(t / HEADS) * 2 * HEADS + HEADS + (t % HEADS)]);
  float rege = ceilf(ranks[n]        * (re / (float)NPAD));
  float regp = ceilf(ranks[NPAD + n] * (rp / (float)NPAD));
  unsigned comb = (unsigned)(rege * (mx + 1.0f) + regp);
  keys[(size_t)a * NS + n] = (comb << 17) | (unsigned)n;
}

__global__ void k_perm(const unsigned* __restrict__ keys, int* __restrict__ perm) {
  int i = blockIdx.x * blockDim.x + threadIdx.x;   // 0 .. 8*NPAD-1
  if (i >= NHASH * HEADS * NPAD) return;
  int a = i / NPAD, pos = i - a * NPAD;
  perm[i] = (int)(keys[(size_t)a * NS + pos] & 0x1FFFFu);
}

// w_h[l][h] = mean(|rpe_w[l]| over (NWPD, HD))
__global__ void k_wh(const float* __restrict__ rpe_w, float* __restrict__ wh) {
  int t = threadIdx.x;
  if (t >= LAYERS * HEADS) return;
  int l = t / HEADS, h = t % HEADS;
  const float* w = rpe_w + (size_t)l * 8 * HDIM;
  float s = 0.f;
  for (int i = 0; i < 8; i++)
    for (int d = 0; d < HD; d++) s += fabsf(w[(size_t)i * HDIM + h * HD + d]);
  wh[t] = s / (8.0f * HD);
}

__global__ void k_zero(float* __restrict__ p, size_t n) {
  size_t i = (size_t)blockIdx.x * blockDim.x + threadIdx.x;
  if (i < n) p[i] = 0.f;
}

// ---------------------------------------------------------------------------
// Generic WMMA GEMM: Y[M,Dout] = act(xscale*X[M,Din] @ W[Din,Dout] + b [+ R])
// wpb waves per block share one LDS A-tile (16x32 f32, b128-staged); each wave
// owns one 16x16 output tile per K-chunk. f32 -> f16 at fragment build,
// f32 accumulate in WMMA.
// ---------------------------------------------------------------------------
__global__ void __launch_bounds__(256) k_gemm(
    const float* __restrict__ X, int ldx,
    const float* __restrict__ Wt,            // row-major [Din, Dout]
    const float* __restrict__ bias,          // may be null
    const float* __restrict__ R, int ldr,    // residual, may be null
    void* __restrict__ Yv, int ldy,
    int Din, int Dout, int act, float xscale, int mvalid, int out_f16) {
  __shared__ float As[16 * 32];              // shared by all waves
  __shared__ float Bs[8 * 32 * 16];          // per-wave 32x16 slices
  int lane = threadIdx.x & 31, w = threadIdx.x >> 5;
  int nw = blockDim.x >> 5;
  int row0 = blockIdx.x * 16;
  int col0 = (blockIdx.y * nw + w) * 16;
  float* Bw = Bs + w * (32 * 16);
  int mg = lane & 15, hg = lane >> 4;
  v8f acc = {};
  for (int k0 = 0; k0 < Din; k0 += 32) {
    if (threadIdx.x == 0 && k0 + 32 < Din)   // hint next W chunk (global_prefetch_b8)
      __builtin_prefetch(Wt + (size_t)(k0 + 32) * Dout + blockIdx.y * nw * 16, 0, 1);
    // A tile: 16x32 f32 = 128 float4, cooperative across the block
    for (int i = threadIdx.x; i < 128; i += blockDim.x) {
      int r = i >> 3, c4 = i & 7;
      float4 vv = make_float4(0.f, 0.f, 0.f, 0.f);
      if (row0 + r < mvalid)
        vv = *(const float4*)(X + (size_t)(row0 + r) * ldx + k0 + c4 * 4);
      vv.x *= xscale; vv.y *= xscale; vv.z *= xscale; vv.w *= xscale;
      *(float4*)&As[r * 32 + c4 * 4] = vv;
    }
    // B tile: 32x16 f32 per wave = 128 float4 / 32 lanes = 4 each
#pragma unroll
    for (int it = 0; it < 4; it++) {
      int i = it * 32 + lane;
      int r = i >> 2, c4 = i & 3;
      *(float4*)&Bw[r * 16 + c4 * 4] =
          *(const float4*)(Wt + (size_t)(k0 + r) * Dout + col0 + c4 * 4);
    }
    __syncthreads();
    v16h av, bv;
#pragma unroll
    for (int e = 0; e < 16; e++) {
      int ka = (e & 7) + ((e >> 3) << 4) + (hg << 3);     // A 16x32 f16 layout
      av[e] = (_Float16)As[mg * 32 + ka];
      int kb = (e & 7) + ((e >> 3) << 4) + (hg << 3);     // B 32x16 f16 layout
      bv[e] = (_Float16)Bw[kb * 16 + mg];
    }
    acc = __builtin_amdgcn_wmma_f32_16x16x32_f16(false, av, false, bv,
                                                 (short)0, acc, false, false);
    __syncthreads();
  }
#pragma unroll
  for (int r = 0; r < 8; r++) {
    int row = row0 + r + hg * 8;
    int col = col0 + mg;
    float v = acc[r];
    if (bias) v += bias[col];
    if (R)    v += R[(size_t)row * ldr + col];
    if (act == 1) v = fmaxf(v, 0.f);
    if (out_f16) ((_Float16*)Yv)[(size_t)row * ldy + col] = (_Float16)v;
    else         ((float*)Yv)[(size_t)row * ldy + col]     = v;
  }
}

// ---------------------------------------------------------------------------
// LayerNorm (one wave per row), optional tanh. Supports D in {64, 256}.
// ---------------------------------------------------------------------------
__global__ void __launch_bounds__(32) k_ln(const float* __restrict__ X, int ldx,
                                           float* __restrict__ Y, int ldy,
                                           const float* __restrict__ g,
                                           const float* __restrict__ b,
                                           int D, int act_tanh) {
  int n = blockIdx.x, lane = threadIdx.x;
  const float* x = X + (size_t)n * ldx;
  float vals[8];
  int per = D / 32;
  float s = 0.f;
  for (int i = 0; i < per; i++) { vals[i] = x[i * 32 + lane]; s += vals[i]; }
  for (int o = 16; o >= 1; o >>= 1) s += __shfl_xor(s, o, 32);
  float mean = s / D;
  float vs = 0.f;
  for (int i = 0; i < per; i++) { float d = vals[i] - mean; vs += d * d; }
  for (int o = 16; o >= 1; o >>= 1) vs += __shfl_xor(vs, o, 32);
  float inv = rsqrtf(vs / D + 1e-5f);
  float* y = Y + (size_t)n * ldy;
  for (int i = 0; i < per; i++) {
    float v = (vals[i] - mean) * inv * g[i * 32 + lane] + b[i * 32 + lane];
    if (act_tanh) v = tanhf(v);
    y[i * 32 + lane] = v;
  }
}

// ---------------------------------------------------------------------------
// Fused block attention: one WG (8 waves) per (block, head, hash).
// Gather Q/K/V (f16, b128 vectorized) + coords + valid via perm into LDS;
// S = QK^T/8 - w_h*d2, mask, softmax (register-resident, shfl row
// reductions); P (f16, overlays Q/K LDS); O = P@V; atomicAdd scatter
// accumulating both hash rounds (2-way commutative add => deterministic).
// ---------------------------------------------------------------------------
__global__ void __launch_bounds__(256) k_attn(
    const _Float16* __restrict__ qg, const _Float16* __restrict__ kg,
    const _Float16* __restrict__ vg, const float* __restrict__ coords,
    const int* __restrict__ perm, const float* __restrict__ wh_all,
    int layer, float* __restrict__ aggr) {
  __shared__ __align__(16) unsigned char smem[3 * 16384 + 128 * 2 * 4 + 128 * 4];
  _Float16* Q16 = (_Float16*)smem;                  // [128][64] (16 KB)
  _Float16* K16 = (_Float16*)(smem + 16384);        // [128][64]
  _Float16* V16 = (_Float16*)(smem + 32768);        // [128][64]
  float*    Cs  = (float*)(smem + 49152);           // [128][2]
  float*    Vld = (float*)(smem + 49152 + 1024);    // [128] valid flags
  _Float16* P16 = (_Float16*)smem;                  // [128][128] overlays Q16+K16

  int blk = blockIdx.x, head = blockIdx.y, hsh = blockIdx.z;
  int t = threadIdx.x;
  const int* pm = perm + ((size_t)(hsh * HEADS + head)) * NPAD + (size_t)blk * 128;

  {  // gather: 2 threads per row, 4x uint4 (64B) each of Q/K/V
    int rowi = t >> 1, half = t & 1;
    int orig = pm[rowi];
    size_t src = (size_t)orig * HDIM + head * HD + half * 32;
    int dst = rowi * 64 + half * 32;
    const uint4* qr = (const uint4*)(qg + src);
    const uint4* kr = (const uint4*)(kg + src);
    const uint4* vr = (const uint4*)(vg + src);
    uint4* qd = (uint4*)(Q16 + dst);
    uint4* kd = (uint4*)(K16 + dst);
    uint4* vd = (uint4*)(V16 + dst);
#pragma unroll
    for (int j = 0; j < 4; j++) { qd[j] = qr[j]; kd[j] = kr[j]; vd[j] = vr[j]; }
    if (half == 0) {
      bool ok = orig < N0V;
      Vld[rowi] = ok ? 1.f : 0.f;
      Cs[rowi * 2 + 0] = ok ? coords[(size_t)orig * 2 + 0] : 0.f;  // pe: pad -> 0
      Cs[rowi * 2 + 1] = ok ? coords[(size_t)orig * 2 + 1] : 0.f;
    }
  }
  __syncthreads();

  int lane = t & 31, w = t >> 5;
  int mg = lane & 15, hg = lane >> 4;
  float wh = wh_all[layer * HEADS + head];

  // ---- S = Q K^T : each wave owns 16 query rows x 128 key cols (8 tiles) ----
  v8f S[8];
#pragma unroll
  for (int tc = 0; tc < 8; tc++) {
    v8f acc = {};
#pragma unroll
    for (int d = 0; d < 64; d += 32) {
      v16h av, bv;
#pragma unroll
      for (int e = 0; e < 16; e++) {
        int ka = (e & 7) + ((e >> 3) << 4) + (hg << 3);
        av[e] = Q16[(w * 16 + mg) * 64 + d + ka];
        int kb = (e & 7) + ((e >> 3) << 4) + (hg << 3);
        bv[e] = K16[(tc * 16 + mg) * 64 + d + kb];       // B[k][n] = K[n][k]
      }
      acc = __builtin_amdgcn_wmma_f32_16x16x32_f16(false, av, false, bv,
                                                   (short)0, acc, false, false);
    }
    S[tc] = acc;
  }

  // ---- RPE + mask + softmax (rows live across lanes 0..15 / 16..31) ----
#pragma unroll
  for (int r = 0; r < 8; r++) {
    int row = w * 16 + r + hg * 8;
    float cx = Cs[row * 2], cy = Cs[row * 2 + 1];
    float m = -3.4e38f;
#pragma unroll
    for (int tc = 0; tc < 8; tc++) {
      int col = tc * 16 + mg;
      float dx = cx - Cs[col * 2], dy = cy - Cs[col * 2 + 1];
      float lg = S[tc][r] * 0.125f - wh * (dx * dx + dy * dy);
      lg = (Vld[col] != 0.f) ? lg : -1e9f;
      S[tc][r] = lg;
      m = fmaxf(m, lg);
    }
    for (int o = 8; o >= 1; o >>= 1) m = fmaxf(m, __shfl_xor(m, o, 32));
    float sum = 0.f;
#pragma unroll
    for (int tc = 0; tc < 8; tc++) {
      float e = __expf(S[tc][r] - m);
      S[tc][r] = e; sum += e;
    }
    for (int o = 8; o >= 1; o >>= 1) sum += __shfl_xor(sum, o, 32);
    float inv = 1.0f / sum;
#pragma unroll
    for (int tc = 0; tc < 8; tc++) S[tc][r] *= inv;
  }

  __syncthreads();  // all waves done reading Q16/K16 before P16 overlays them
#pragma unroll
  for (int tc = 0; tc < 8; tc++)
#pragma unroll
    for (int r = 0; r < 8; r++) {
      int row = w * 16 + r + hg * 8, col = tc * 16 + mg;
      P16[row * 128 + col] = (_Float16)S[tc][r];
    }
  __syncthreads();

  // ---- O = P @ V : 4 output tiles per wave, K-loop over 128 ----
#pragma unroll
  for (int oc = 0; oc < 4; oc++) {
    v8f acc = {};
#pragma unroll
    for (int kk = 0; kk < 128; kk += 32) {
      v16h av, bv;
#pragma unroll
      for (int e = 0; e < 16; e++) {
        int ka = (e & 7) + ((e >> 3) << 4) + (hg << 3);
        av[e] = P16[(w * 16 + mg) * 128 + kk + ka];
        int kb = (e & 7) + ((e >> 3) << 4) + (hg << 3);
        bv[e] = V16[(kk + kb) * 64 + oc * 16 + mg];
      }
      acc = __builtin_amdgcn_wmma_f32_16x16x32_f16(false, av, false, bv,
                                                   (short)0, acc, false, false);
    }
#pragma unroll
    for (int r = 0; r < 8; r++) {
      int row = w * 16 + r + hg * 8, col = oc * 16 + mg;
      int orig = pm[row];
      atomicAdd(&aggr[(size_t)orig * HDIM + head * HD + col], acc[r]);
    }
  }
}

// ---------------------------------------------------------------------------
// Final head: out = (enc + m)[:N0] @ out_w + out_b
// ---------------------------------------------------------------------------
__global__ void k_out(const float* __restrict__ enc, const float* __restrict__ m,
                      const float* __restrict__ ow, const float* __restrict__ ob,
                      float* __restrict__ out) {
  int n = blockIdx.x * blockDim.x + threadIdx.x;
  if (n >= N0V) return;
  float acc[4] = {ob[0], ob[1], ob[2], ob[3]};
  for (int i = 0; i < D2; i++) {
    float v = enc[(size_t)n * D2 + i] + m[(size_t)n * D2 + i];
    for (int c = 0; c < 4; c++) acc[c] += v * ow[i * 4 + c];
  }
  for (int c = 0; c < 4; c++) out[(size_t)n * 4 + c] = acc[c];
}

// ---------------------------------------------------------------------------
// Host orchestration
// ---------------------------------------------------------------------------
extern "C" void kernel_launch(void* const* d_in, const int* in_sizes, int n_in,
                              void* d_out, int out_size, void* d_ws, size_t ws_size,
                              hipStream_t stream) {
  (void)in_sizes; (void)n_in; (void)out_size; (void)ws_size;

  int ii = 0;
  const float* x       = (const float*)d_in[ii++];   // 0
  const float* coords  = (const float*)d_in[ii++];   // 1
  ii++;                                              // 2 batch (unused)
  const float* regions = (const float*)d_in[ii++];   // 3
  const float* feat_w1 = (const float*)d_in[ii++];
  const float* feat_b1 = (const float*)d_in[ii++];
  const float* feat_w2 = (const float*)d_in[ii++];
  const float* feat_b2 = (const float*)d_in[ii++];
  const float* ln1_g   = (const float*)d_in[ii++];
  const float* ln1_b   = (const float*)d_in[ii++];
  const float* wq      = (const float*)d_in[ii++];
  const float* wk      = (const float*)d_in[ii++];
  const float* wv      = (const float*)d_in[ii++];
  const float* rpe_w   = (const float*)d_in[ii++];
  ii++;                                              // rpe_b (unused in fwd)
  const float* wo_w    = (const float*)d_in[ii++];
  const float* wo_b    = (const float*)d_in[ii++];
  const float* ln2_g   = (const float*)d_in[ii++];
  const float* ln2_b   = (const float*)d_in[ii++];
  const float* ff_w1   = (const float*)d_in[ii++];
  const float* ff_b1   = (const float*)d_in[ii++];
  const float* ff_w2   = (const float*)d_in[ii++];
  const float* ff_b2   = (const float*)d_in[ii++];
  const float* Wenc    = (const float*)d_in[ii++];
  const float* mlp_ws[5]; for (int j = 0; j < 5; j++) mlp_ws[j] = (const float*)d_in[ii++];
  const float* mlp_bs[5]; for (int j = 0; j < 5; j++) mlp_bs[j] = (const float*)d_in[ii++];
  const float* mlp_ng[4]; for (int j = 0; j < 4; j++) mlp_ng[j] = (const float*)d_in[ii++];
  const float* mlp_nb[4]; for (int j = 0; j < 4; j++) mlp_nb[j] = (const float*)d_in[ii++];
  const float* out_w   = (const float*)d_in[ii++];
  const float* out_b   = (const float*)d_in[ii++];

  // workspace arena
  unsigned char* base = (unsigned char*)d_ws;
  size_t off = 0;
  auto arena = [&](size_t bytes) -> void* {
    void* p = base + off;
    off += (bytes + 255) & ~(size_t)255;
    return p;
  };
  unsigned long long* keys64 = (unsigned long long*)arena(2ull * NS * 8);
  float*     ranks  = (float*)arena(2ull * NPAD * 4);
  unsigned*  keysc  = (unsigned*)arena((size_t)NHASH * HEADS * NS * 4);
  int*       perm   = (int*)arena((size_t)NHASH * HEADS * NPAD * 4);
  float*     wh     = (float*)arena(64);
  float*     outsb  = (float*)arena((size_t)NPAD * ENCD * 4);
  float*     hn     = (float*)arena((size_t)NPAD * 64 * 4);
  float*     t64    = (float*)arena((size_t)NPAD * 64 * 4);
  _Float16*  qb     = (_Float16*)arena((size_t)NPAD * HDIM * 2);  // qb+kb contiguous
  _Float16*  kb     = (_Float16*)arena((size_t)NPAD * HDIM * 2);
  _Float16*  vb     = (_Float16*)arena((size_t)NPAD * HDIM * 2);
  float*     aggr   = (float*)arena((size_t)NPAD * HDIM * 4);
  float*     t256b  = (float*)arena((size_t)NPAD * MLPH * 4);
  float*     mbuf   = (float*)arena((size_t)NPAD * D2 * 4);
  float*     t256a  = (float*)qb;   // overlays qb+kb (2*NPAD*256*2 = NPAD*256*4 B)
  float*     encb   = (float*)vb;   // overlays vb (needs NPAD*32*4 <= NPAD*256*2)
  (void)kb;

  auto gemm = [&](const float* X, int ldx, const float* Wt, const float* bias,
                  const float* R, int ldr, void* Y, int ldy,
                  int Din, int Dout, int act, float xscale, int mvalid, int f16o) {
    int wpb = Dout / 16; if (wpb > 8) wpb = 8;         // waves per block
    dim3 g(NPAD / 16, Dout / (16 * wpb));
    k_gemm<<<g, 32 * wpb, 0, stream>>>(X, ldx, Wt, bias, R, ldr, Y, ldy,
                                       Din, Dout, act, xscale, mvalid, f16o);
  };

  // ---- region partition: stable argsorts via bitonic networks ----
  k_keys_coords<<<(2 * NS + 255) / 256, 256, 0, stream>>>(coords, keys64);
  for (int kk = 2; kk <= NS; kk <<= 1)
    for (int jj = kk >> 1; jj >= 1; jj >>= 1)
      k_bitonic64<<<dim3(NS / 256, 2), 256, 0, stream>>>(keys64, kk, jj);
  k_ranks<<<(2 * NS + 255) / 256, 256, 0, stream>>>(keys64, ranks);
  k_combkeys<<<(NHASH * HEADS * NS + 255) / 256, 256, 0, stream>>>(ranks, regions, keysc);
  for (int kk = 2; kk <= NS; kk <<= 1)
    for (int jj = kk >> 1; jj >= 1; jj >>= 1)
      k_bitonic32<<<dim3(NS / 256, NHASH * HEADS), 256, 0, stream>>>(keysc, kk, jj);
  k_perm<<<(NHASH * HEADS * NPAD + 255) / 256, 256, 0, stream>>>(keysc, perm);
  k_wh<<<1, 32, 0, stream>>>(rpe_w, wh);

  // ---- feature encoder: h0 = relu(x@W1+b1)@W2+b2 ----
  gemm(x, INDIM, feat_w1, feat_b1, nullptr, 0, t64, 64, INDIM, 64, 1, 1.f, N0V, 0);
  gemm(t64, 64, feat_w2, feat_b2, nullptr, 0, outsb, ENCD, 64, 64, 0, 1.f, NPAD, 0);

  // ---- transformer layers ----
  for (int l = 0; l < LAYERS; l++) {
    float* hcur  = outsb + l * 64;        // column slice, ld = ENCD
    float* hnext = outsb + (l + 1) * 64;
    k_ln<<<NPAD, 32, 0, stream>>>(hcur, ENCD, hn, 64, ln1_g + l * 64, ln1_b + l * 64, 64, 0);
    gemm(hn, 64, wq + (size_t)l * 64 * HDIM, nullptr, nullptr, 0, qb, HDIM, 64, HDIM, 0, 1.f, NPAD, 1);
    gemm(hn, 64, wk + (size_t)l * 64 * HDIM, nullptr, nullptr, 0, kb, HDIM, 64, HDIM, 0, 1.f, NPAD, 1);
    gemm(hn, 64, wv + (size_t)l * 64 * HDIM, nullptr, nullptr, 0, vb, HDIM, 64, HDIM, 0, 1.f, NPAD, 1);
    k_zero<<<(int)(((size_t)NPAD * HDIM + 255) / 256), 256, 0, stream>>>(aggr, (size_t)NPAD * HDIM);
    k_attn<<<dim3(NBLK, HEADS, NHASH), 256, 0, stream>>>(qb, kb, vb, coords, perm, wh, l, aggr);
    // h = h + mean_hash(aggr) @ wo + b   (0.5 folded into A conversion)
    gemm(aggr, HDIM, wo_w + (size_t)l * HDIM * 64, wo_b + l * 64, hcur, ENCD,
         hnext, ENCD, HDIM, 64, 0, 0.5f, NPAD, 0);
    k_ln<<<NPAD, 32, 0, stream>>>(hnext, ENCD, hn, 64, ln2_g + l * 64, ln2_b + l * 64, 64, 0);
    gemm(hn, 64, ff_w1 + (size_t)l * 64 * 64, ff_b1 + l * 64, nullptr, 0, t64, 64, 64, 64, 1, 1.f, NPAD, 0);
    gemm(t64, 64, ff_w2 + (size_t)l * 64 * 64, ff_b2 + l * 64, hnext, ENCD, hnext, ENCD, 64, 64, 0, 1.f, NPAD, 0);
  }

  // ---- enc = concat(outs) @ W ----
  gemm(outsb, ENCD, Wenc, nullptr, nullptr, 0, encb, D2, ENCD, D2, 0, 1.f, NPAD, 0);

  // ---- MLP: 4x (lin -> LN -> tanh), final lin ----
  gemm(encb, D2, mlp_ws[0], mlp_bs[0], nullptr, 0, t256a, MLPH, D2, MLPH, 0, 1.f, NPAD, 0);
  k_ln<<<NPAD, 32, 0, stream>>>(t256a, MLPH, t256a, MLPH, mlp_ng[0], mlp_nb[0], MLPH, 1);
  gemm(t256a, MLPH, mlp_ws[1], mlp_bs[1], nullptr, 0, t256b, MLPH, MLPH, MLPH, 0, 1.f, NPAD, 0);
  k_ln<<<NPAD, 32, 0, stream>>>(t256b, MLPH, t256b, MLPH, mlp_ng[1], mlp_nb[1], MLPH, 1);
  gemm(t256b, MLPH, mlp_ws[2], mlp_bs[2], nullptr, 0, t256a, MLPH, MLPH, MLPH, 0, 1.f, NPAD, 0);
  k_ln<<<NPAD, 32, 0, stream>>>(t256a, MLPH, t256a, MLPH, mlp_ng[2], mlp_nb[2], MLPH, 1);
  gemm(t256a, MLPH, mlp_ws[3], mlp_bs[3], nullptr, 0, t256b, MLPH, MLPH, MLPH, 0, 1.f, NPAD, 0);
  k_ln<<<NPAD, 32, 0, stream>>>(t256b, MLPH, t256b, MLPH, mlp_ng[3], mlp_nb[3], MLPH, 1);
  gemm(t256b, MLPH, mlp_ws[4], mlp_bs[4], nullptr, 0, mbuf, D2, MLPH, D2, 0, 1.f, NPAD, 0);

  // ---- output head ----
  k_out<<<(N0V + 255) / 256, 256, 0, stream>>>(encb, mbuf, out_w, out_b, (float*)d_out);
}
